// PtrNet_14113262535166
// MI455X (gfx1250) — matrix operand order repaired and checked by reading, not compiled, not run on hardware
//
#include <hip/hip_runtime.h>
#include <hip/hip_bf16.h>

// ---------------------------------------------------------------------------
// PtrNet (encoder LSTM -> decoder LSTM -> additive attention -> softmax)
// B=32, T_E=T_D=128, H=256. All matmuls via v_wmma_f32_16x16x32_bf16.
// Round 2: bounded-pressure K loops (no scratch spills in the LSTM recurrence).
// ---------------------------------------------------------------------------

typedef __attribute__((ext_vector_type(16))) __bf16 v16bf;
typedef __attribute__((ext_vector_type(8)))  float  v8f;

union Frag16 { v16bf v; uint4 q[2]; };

#define HID   256
#define BATCH 32
#define TLEN  128

__device__ __forceinline__ unsigned short f2bf(float f) {
  unsigned u = __float_as_uint(f);
  u += 0x7FFFu + ((u >> 16) & 1u);          // round-to-nearest-even
  return (unsigned short)(u >> 16);
}
__device__ __forceinline__ float sigm(float x) {
  return 1.0f / (1.0f + __expf(-x));
}
__device__ __forceinline__ float tanh_fast(float x) {
  float e = __expf(2.0f * x);               // saturates to +/-1 correctly
  return 1.0f - 2.0f / (e + 1.0f);
}

// --------------------------- fp32 -> bf16 convert ---------------------------
__global__ void f32_to_bf16_kernel(const float* __restrict__ src,
                                   unsigned short* __restrict__ dst, int n) {
  int i = blockIdx.x * blockDim.x + threadIdx.x;
  if (i < n) dst[i] = f2bf(src[i]);
}

// ------------------------------- LSTM scan ----------------------------------
// One workgroup (16 waves). Wave w owns hidden slice hh in [16w, 16w+16) and
// computes the i/f/g/o gate tiles for that slice, so c stays in registers and
// only h (bf16) goes through LDS each step. Weights are streamed from L2 every
// step (512 KB bf16, L2-resident); the kc loop is unrolled only 2x so the
// register allocator never spills in the sequential hot loop.
__global__ __launch_bounds__(512) void lstm_wmma_kernel(
    const float* __restrict__ x, int T,                 // [B, T]
    const unsigned short* __restrict__ Whh_bf,          // [4H, H] bf16
    const float* __restrict__ Wih,                      // [4H]
    const float* __restrict__ bih,
    const float* __restrict__ bhh,                      // [4H]
    const float* __restrict__ h0,                       // [B,H] or null
    const float* __restrict__ c0,                       // [B,H] or null
    const int*   __restrict__ seqm,                     // [B] or null
    unsigned short* __restrict__ hs_bf,                 // [B*T, H] bf16 out
    float* __restrict__ cap_h, float* __restrict__ cap_c) // [B,H] (if seqm)
{
  __shared__ unsigned short hlds[BATCH * HID];          // 16 KB

  const int tid  = threadIdx.x;
  const int lane = tid & 31;
  const int wv   = tid >> 5;                            // 0..15
  const int half = lane >> 4;
  const int lm   = lane & 15;
  const int hh   = wv * 16 + lm;                        // hidden col 0..255

  for (int idx = tid; idx < BATCH * HID; idx += 512)
    hlds[idx] = h0 ? f2bf(h0[idx]) : (unsigned short)0;

  float creg[2][8];
  int   blast[2][8];
#pragma unroll
  for (int mt = 0; mt < 2; ++mt)
#pragma unroll
    for (int r = 0; r < 8; ++r) {
      int b = mt * 16 + half * 8 + r;
      creg[mt][r]  = c0 ? c0[b * HID + hh] : 0.0f;
      blast[mt][r] = seqm ? (seqm[b] - 1) : -1;
    }

  float wih[4], bias[4];
#pragma unroll
  for (int g = 0; g < 4; ++g) {
    wih[g]  = Wih[g * HID + hh];
    bias[g] = bih[g * HID + hh] + bhh[g * HID + hh];
  }
  __syncthreads();

  for (int t = 0; t < T; ++t) {
    v8f acc[2][4];
#pragma unroll
    for (int mt = 0; mt < 2; ++mt)
#pragma unroll
      for (int g = 0; g < 4; ++g)
#pragma unroll
        for (int r = 0; r < 8; ++r) acc[mt][g][r] = 0.0f;

    // K loop kept as a runtime loop (unroll 2): addresses depend on kc, so
    // LICM cannot hoist the 256-VGPR weight working set out of the t-loop.
#pragma unroll 2
    for (int kc = 0; kc < 8; ++kc) {
      // A fragments for both M tiles (16x32 bf16, contiguous k order used here:
      // row = lane%16, VGPR0..7 -> k = half*16+{0..15}).
      Frag16 a0, a1;
      {
        const uint4* ap0 =
            (const uint4*)&hlds[lm * HID + kc * 32 + half * 16];
        const uint4* ap1 =
            (const uint4*)&hlds[(16 + lm) * HID + kc * 32 + half * 16];
        a0.q[0] = ap0[0];
        a0.q[1] = ap0[1];
        a1.q[0] = ap1[0];
        a1.q[1] = ap1[1];
      }
#pragma unroll
      for (int g = 0; g < 4; ++g) {
        // B fragment: 32x16 bf16, col = lane%16, lanes<16 hold K 0..15,
        // lanes>=16 hold K 16..31 -> 16 contiguous bf16 per lane.
        // Shared by both M tiles.
        const int n = g * HID + hh;
        const uint4* bp =
            (const uint4*)&Whh_bf[n * HID + kc * 32 + half * 16];
        Frag16 bm;
        bm.q[0] = bp[0];
        bm.q[1] = bp[1];
        acc[0][g] = __builtin_amdgcn_wmma_f32_16x16x32_bf16(
            false, a0.v, false, bm.v, (short)0, acc[0][g], false, false);
        acc[1][g] = __builtin_amdgcn_wmma_f32_16x16x32_bf16(
            false, a1.v, false, bm.v, (short)0, acc[1][g], false, false);
      }
    }
    __syncthreads();   // all reads of old h complete

#pragma unroll
    for (int mt = 0; mt < 2; ++mt) {
#pragma unroll
      for (int r = 0; r < 8; ++r) {
        const int b   = mt * 16 + half * 8 + r;
        const float xv = x[b * T + t];
        float gi = acc[mt][0][r] + xv * wih[0] + bias[0];
        float gf = acc[mt][1][r] + xv * wih[1] + bias[1];
        float gg = acc[mt][2][r] + xv * wih[2] + bias[2];
        float go = acc[mt][3][r] + xv * wih[3] + bias[3];
        float cn = sigm(gf) * creg[mt][r] + sigm(gi) * tanh_fast(gg);
        float hv = sigm(go) * tanh_fast(cn);
        creg[mt][r] = cn;
        unsigned short hb = f2bf(hv);
        hlds[b * HID + hh]            = hb;
        hs_bf[(b * T + t) * HID + hh] = hb;
        if (t == blast[mt][r]) {
          cap_h[b * HID + hh] = hv;
          cap_c[b * HID + hh] = cn;
        }
      }
    }
    __syncthreads();   // new h visible before next step
  }
}

// --------------------- Y = X @ W^T + bias  (bf16 WMMA) ----------------------
// X: [M,256] bf16, W: [256,256] bf16 (row n = output column), Y: [M,256] f32.
// One block per 16-row tile; 8 waves x 2 N-tiles; K loop unrolled 2x.
__global__ __launch_bounds__(256) void gemm_bf16_kernel(
    const unsigned short* __restrict__ Xbf,
    const unsigned short* __restrict__ Wbf,
    const float* __restrict__ bias,
    float* __restrict__ Y)
{
  const int tid  = threadIdx.x;
  const int lane = tid & 31;
  const int wv   = tid >> 5;             // 0..7
  const int half = lane >> 4;
  const int lm   = lane & 15;
  const int mt   = blockIdx.x;

  v8f acc[2];
#pragma unroll
  for (int j = 0; j < 2; ++j)
#pragma unroll
    for (int r = 0; r < 8; ++r) acc[j][r] = 0.0f;

#pragma unroll 2
  for (int kc = 0; kc < 8; ++kc) {
    Frag16 a;
    const uint4* ap =
        (const uint4*)&Xbf[(mt * 16 + lm) * HID + kc * 32 + half * 16];
    a.q[0] = ap[0];
    a.q[1] = ap[1];
#pragma unroll
    for (int j = 0; j < 2; ++j) {
      const int n = (wv * 2 + j) * 16 + lm;
      const uint4* bp = (const uint4*)&Wbf[n * HID + kc * 32 + half * 16];
      Frag16 bm;
      bm.q[0] = bp[0];
      bm.q[1] = bp[1];
      acc[j] = __builtin_amdgcn_wmma_f32_16x16x32_bf16(
          false, a.v, false, bm.v, (short)0, acc[j], false, false);
    }
  }
#pragma unroll
  for (int j = 0; j < 2; ++j) {
    const int n = (wv * 2 + j) * 16 + lm;
#pragma unroll
    for (int r = 0; r < 8; ++r) {
      const int row = mt * 16 + half * 8 + r;
      Y[row * HID + n] = acc[j][r] + bias[n];
    }
  }
}

// ----------------- attention logits + mask + softmax (fused) ----------------
// One wave per (b, i); 4 encoder positions j per lane; shuffle reductions.
__global__ __launch_bounds__(256) void attn_softmax_kernel(
    const float* __restrict__ e2,     // [B*T_E, H]
    const float* __restrict__ d2,     // [B*T_D, H]
    const float* __restrict__ vvec,   // [H]
    const float* __restrict__ seq,    // [B, T_E]
    float* __restrict__ out)          // [B, T_D, T_E]
{
  const int tid  = threadIdx.x;
  const int lane = tid & 31;
  const int wv   = tid >> 5;
  const int b    = blockIdx.x >> 4;
  const int i    = (blockIdx.x & 15) * 8 + wv;

  const float4* dd = (const float4*)&d2[(b * TLEN + i) * HID];
  const float4* v4 = (const float4*)vvec;

  float logit[4];
#pragma unroll
  for (int jj = 0; jj < 4; ++jj) {
    const int j = jj * 32 + lane;
    const float4* ee = (const float4*)&e2[(b * TLEN + j) * HID];
    float s = 0.0f;
    for (int h = 0; h < HID / 4; ++h) {
      float4 dv = dd[h], ev = ee[h], vv = v4[h];
      s += tanh_fast(dv.x + ev.x) * vv.x;
      s += tanh_fast(dv.y + ev.y) * vv.y;
      s += tanh_fast(dv.z + ev.z) * vv.z;
      s += tanh_fast(dv.w + ev.w) * vv.w;
    }
    // mask: seq with [:,0]:=0.1 -> position 0 always valid; zeros get -1000
    float sv   = seq[b * TLEN + j];
    logit[jj]  = (j == 0 || sv != 0.0f) ? s : s - 1000.0f;
  }

  float mx = fmaxf(fmaxf(logit[0], logit[1]), fmaxf(logit[2], logit[3]));
#pragma unroll
  for (int off = 16; off; off >>= 1) mx = fmaxf(mx, __shfl_xor(mx, off, 32));

  float ex[4], sum = 0.0f;
#pragma unroll
  for (int jj = 0; jj < 4; ++jj) { ex[jj] = __expf(logit[jj] - mx); sum += ex[jj]; }
#pragma unroll
  for (int off = 16; off; off >>= 1) sum += __shfl_xor(sum, off, 32);

  const float inv = 1.0f / sum;
#pragma unroll
  for (int jj = 0; jj < 4; ++jj)
    out[(b * TLEN + i) * TLEN + jj * 32 + lane] = ex[jj] * inv;
}

// ------------------------------- host glue ----------------------------------
extern "C" void kernel_launch(void* const* d_in, const int* in_sizes, int n_in,
                              void* d_out, int out_size, void* d_ws, size_t ws_size,
                              hipStream_t stream) {
  (void)in_sizes; (void)n_in; (void)out_size; (void)ws_size;

  const float* seq    = (const float*)d_in[0];
  const int*   seqm   = (const int*)  d_in[1];
  const float* target = (const float*)d_in[2];
  const float* Wih_e  = (const float*)d_in[3];
  const float* Whh_e  = (const float*)d_in[4];
  const float* bih_e  = (const float*)d_in[5];
  const float* bhh_e  = (const float*)d_in[6];
  const float* Wih_d  = (const float*)d_in[7];
  const float* Whh_d  = (const float*)d_in[8];
  const float* bih_d  = (const float*)d_in[9];
  const float* bhh_d  = (const float*)d_in[10];
  const float* We     = (const float*)d_in[11];
  const float* be     = (const float*)d_in[12];
  const float* Wd     = (const float*)d_in[13];
  const float* bd     = (const float*)d_in[14];
  const float* vvec   = (const float*)d_in[15];

  char* ws = (char*)d_ws;
  auto take = [&](size_t bytes) -> char* {
    char* p = ws;
    ws += (bytes + 255) & ~size_t(255);
    return p;
  };
  unsigned short* Whh_e_bf = (unsigned short*)take(1024 * 256 * 2);
  unsigned short* Whh_d_bf = (unsigned short*)take(1024 * 256 * 2);
  unsigned short* We_bf    = (unsigned short*)take(256 * 256 * 2);
  unsigned short* Wd_bf    = (unsigned short*)take(256 * 256 * 2);
  unsigned short* e_bf     = (unsigned short*)take((size_t)BATCH * TLEN * HID * 2);
  unsigned short* d_bf     = (unsigned short*)take((size_t)BATCH * TLEN * HID * 2);
  float*          eh       = (float*)take(BATCH * HID * 4);
  float*          ec       = (float*)take(BATCH * HID * 4);
  float*          e2       = (float*)take((size_t)BATCH * TLEN * HID * 4);
  float*          d2       = (float*)take((size_t)BATCH * TLEN * HID * 4);

  // 1) weights fp32 -> bf16
  f32_to_bf16_kernel<<<(1024 * 256 + 255) / 256, 256, 0, stream>>>(Whh_e, Whh_e_bf, 1024 * 256);
  f32_to_bf16_kernel<<<(1024 * 256 + 255) / 256, 256, 0, stream>>>(Whh_d, Whh_d_bf, 1024 * 256);
  f32_to_bf16_kernel<<<(256 * 256 + 255) / 256, 256, 0, stream>>>(We, We_bf, 256 * 256);
  f32_to_bf16_kernel<<<(256 * 256 + 255) / 256, 256, 0, stream>>>(Wd, Wd_bf, 256 * 256);

  // 2) encoder scan (captures h/c at seq_m-1)
  lstm_wmma_kernel<<<1, 512, 0, stream>>>(seq, TLEN, Whh_e_bf, Wih_e, bih_e, bhh_e,
                                          nullptr, nullptr, seqm, e_bf, eh, ec);
  // 3) decoder scan (init from captured encoder state)
  lstm_wmma_kernel<<<1, 512, 0, stream>>>(target, TLEN, Whh_d_bf, Wih_d, bih_d, bhh_d,
                                          eh, ec, nullptr, d_bf, eh, ec);

  // 4) e2 = e @ We^T + be ; d2 = d @ Wd^T + bd
  gemm_bf16_kernel<<<BATCH * TLEN / 16, 256, 0, stream>>>(e_bf, We_bf, be, e2);
  gemm_bf16_kernel<<<BATCH * TLEN / 16, 256, 0, stream>>>(d_bf, Wd_bf, bd, d2);

  // 5) attention logits + mask + softmax
  attn_softmax_kernel<<<BATCH * (TLEN / 8), 256, 0, stream>>>(e2, d2, vvec, seq,
                                                              (float*)d_out);
}
